// Conv2DMod_71871982731629
// MI455X (gfx1250) — matrix operand order repaired and checked by reading, not compiled, not run on hardware
//
#include <hip/hip_runtime.h>
#include <hip/hip_bf16.h>

typedef __bf16 bf16_t;
typedef __attribute__((ext_vector_type(16))) __bf16 v16bf;
typedef __attribute__((ext_vector_type(8)))  __bf16 v8bf;
typedef __attribute__((ext_vector_type(8)))  float  v8f;

#define BATCH 8
#define CIN   256
#define COUT  256
#define HW    4096                 // 64*64
#define PW    66                   // padded width/height
#define PHW   (PW*PW)              // 4356 padded spatial positions
#define WBF_ELEMS (9*COUT*CIN)     // 589824 bf16
#define WBF_BYTES (WBF_ELEMS*2)    // 1,179,648 B (16B aligned)
#define DMOD_BYTES (BATCH*COUT*4)  // 8192 B
#define XP_ELEMS ((size_t)BATCH*PHW*CIN)  // 8,921,088 bf16
#define XP_BYTES (XP_ELEMS*2)             // 17,842,176 B

// ---------------- Kernel 0: zero the padded activation buffer ---------------
__global__ void zero_ws_kernel(uint4* __restrict__ p, int n16) {
    int t = blockIdx.x * 256 + threadIdx.x;
    if (t < n16) p[t] = make_uint4(0u, 0u, 0u, 0u);
}

// ---------------- Kernel 1: weight fp32 -> bf16, layout [kpos][o][ci] -------
__global__ void wconv_kernel(const float* __restrict__ w, bf16_t* __restrict__ wbf) {
    int t = blockIdx.x * 256 + threadIdx.x;     // 0 .. 589823
    int i    = t & 255;
    int o    = (t >> 8) & 255;
    int kpos = t >> 16;                          // 0..8
    float v = w[((size_t)o * CIN + i) * 9 + kpos];
    wbf[t] = (__bf16)v;
}

// ---------------- Kernel 2: demod factor d[b][o] ----------------------------
__global__ void modnorm_kernel(const float* __restrict__ w, const float* __restrict__ y,
                               float* __restrict__ dmod) {
    int o = blockIdx.x;
    int i = threadIdx.x;
    const float* wp = w + ((size_t)o * CIN + i) * 9;
    float s2 = 0.f;
#pragma unroll
    for (int k = 0; k < 9; ++k) { float v = wp[k]; s2 += v * v; }
    __shared__ float red[256];
    for (int b = 0; b < BATCH; ++b) {
        float m = y[b * CIN + i] + 1.0f;
        red[i] = s2 * m * m;
        __syncthreads();
        for (int s = 128; s > 0; s >>= 1) {
            if (i < s) red[i] += red[i + s];
            __syncthreads();
        }
        if (i == 0) dmod[b * COUT + o] = rsqrtf(red[0] + 1e-6f);
        __syncthreads();
    }
}

// ---- Kernel 3: x fp32 [b][c][p] -> modulated bf16 xp[b][py+1][px+1][c] -----
__global__ void xmod_transpose_kernel(const float* __restrict__ x, const float* __restrict__ y,
                                      bf16_t* __restrict__ xp) {
    __shared__ __bf16 tile[32][33];
    int b  = blockIdx.z;
    int c0 = blockIdx.x * 32;
    int p0 = blockIdx.y * 32;
    int tx = threadIdx.x & 31;
    int ty = threadIdx.x >> 5;   // 0..7
#pragma unroll
    for (int i = 0; i < 4; ++i) {
        int cl = ty + i * 8;
        float m = y[b * CIN + c0 + cl] + 1.0f;
        float v = x[(((size_t)b * CIN + c0 + cl) * HW) + p0 + tx];
        tile[cl][tx] = (__bf16)(v * m);
    }
    __syncthreads();
#pragma unroll
    for (int i = 0; i < 4; ++i) {
        int pl  = ty + i * 8;
        int p   = p0 + pl;
        int row = p >> 6;
        int col = p & 63;
        xp[(((size_t)b * PHW) + (row + 1) * PW + (col + 1)) * CIN + c0 + tx] = tile[tx][pl];
    }
}

// ---------------- Kernel 4: implicit GEMM conv (bf16 WMMA, f32 accum) -------
// Grid: (O/128=2, HW/256=16, B=8) = 256 blocks, 256 threads (8 waves).
// Wave (wm in 0..1, wn in 0..3): 4 M-tiles x 4 N-tiles of 16x16 each
// -> 16 WMMAs per k-step fed by 8 A-loads + 8 B-loads (1.0 b128 load / WMMA).
// Fully unrolled kpos(9) x ct(8): all loads use constant immediate offsets
// off 8 base pointers; no branches / no address math in the hot path.
__global__ __launch_bounds__(256) void conv_gemm_kernel(
        const bf16_t* __restrict__ wbf, const bf16_t* __restrict__ xp,
        const float* __restrict__ dmod, float* __restrict__ out) {
    int b    = blockIdx.z;
    int tid  = threadIdx.x;
    int lane = tid & 31;
    int wv   = tid >> 5;
    int wm   = wv & 1;
    int wn   = wv >> 1;
    int lm   = lane & 15;
    int hsel = lane >> 4;          // 0: lanes 0-15, 1: lanes 16-31
    int m_base = blockIdx.x * 128 + wm * 64;
    int n_base = blockIdx.y * 256 + wn * 64;

    // A row base pointers (per M-tile): lane = row; lanes<16 hold K 0-15,
    // lanes>=16 hold K 16-31 (second chunk at +8 elements).
    const bf16_t* abase[4];
#pragma unroll
    for (int tm = 0; tm < 4; ++tm)
        abase[tm] = wbf + (size_t)(m_base + tm * 16 + lm) * CIN + hsel * 16;

    // B column base pointers (per N-tile): lane = col; 16 consecutive channels,
    // lanes<16 hold K 0-15, lanes>=16 hold K 16-31. Centered in padded grid.
    const bf16_t* bbase[4];
#pragma unroll
    for (int tn = 0; tn < 4; ++tn) {
        int n  = n_base + tn * 16 + lm;
        int py = n >> 6;
        int px = n & 63;
        bbase[tn] = xp + (((size_t)b * PHW) + (size_t)(py + 1) * PW + (px + 1)) * CIN
                    + hsel * 16;
    }

    // L2 warm-up hints
    __builtin_prefetch(abase[0], 0, 1);
    __builtin_prefetch(bbase[0], 0, 1);

    v8f acc[4][4];
#pragma unroll
    for (int tm = 0; tm < 4; ++tm)
#pragma unroll
        for (int tn = 0; tn < 4; ++tn)
#pragma unroll
            for (int e = 0; e < 8; ++e) acc[tm][tn][e] = 0.0f;

    union U16 { v16bf v; v8bf h[2]; };

#pragma unroll
    for (int kpos = 0; kpos < 9; ++kpos) {
        const int dy   = kpos / 3 - 1;
        const int dx   = kpos % 3 - 1;
        const int aoff = kpos * (COUT * CIN);      // elements; const per unroll
        const int boff = (dy * PW + dx) * CIN;     // elements; const per unroll
#pragma unroll
        for (int ct = 0; ct < 8; ++ct) {
            const int cb = ct * 32;
            v16bf amat[4];
#pragma unroll
            for (int tm = 0; tm < 4; ++tm) {
                U16 ua;
                ua.h[0] = *(const v8bf*)(abase[tm] + aoff + cb);
                ua.h[1] = *(const v8bf*)(abase[tm] + aoff + cb + 8);
                amat[tm] = ua.v;
            }
            v16bf bmat[4];
#pragma unroll
            for (int tn = 0; tn < 4; ++tn) {
                U16 ub;
                ub.h[0] = *(const v8bf*)(bbase[tn] + boff + cb);
                ub.h[1] = *(const v8bf*)(bbase[tn] + boff + cb + 8);
                bmat[tn] = ub.v;
            }
#pragma unroll
            for (int tm = 0; tm < 4; ++tm)
#pragma unroll
                for (int tn = 0; tn < 4; ++tn)
                    acc[tm][tn] = __builtin_amdgcn_wmma_f32_16x16x32_bf16(
                        false, amat[tm], false, bmat[tn],
                        (short)0, acc[tm][tn], false, false);
        }
    }

    // Epilogue: demodulate and store fp32 output [b][o][p]
    float* op = out + (size_t)b * COUT * HW;
#pragma unroll
    for (int tm = 0; tm < 4; ++tm) {
#pragma unroll
        for (int e = 0; e < 8; ++e) {
            int o    = m_base + tm * 16 + hsel * 8 + e;  // C/D layout: VGPR e -> row
            float dv = dmod[b * COUT + o];
#pragma unroll
            for (int tn = 0; tn < 4; ++tn) {
                int n = n_base + tn * 16 + lm;
                op[(size_t)o * HW + n] = acc[tm][tn][e] * dv;
            }
        }
    }
}

extern "C" void kernel_launch(void* const* d_in, const int* in_sizes, int n_in,
                              void* d_out, int out_size, void* d_ws, size_t ws_size,
                              hipStream_t stream) {
    const float* x = (const float*)d_in[0];   // [8,256,64,64]
    const float* y = (const float*)d_in[1];   // [8,256]
    const float* w = (const float*)d_in[2];   // [256,256,3,3]
    float* out = (float*)d_out;

    size_t need = (size_t)WBF_BYTES + DMOD_BYTES + XP_BYTES;
    if (ws_size < need) return;               // not enough scratch; bail deterministically

    char* ws = (char*)d_ws;
    bf16_t* wbf  = (bf16_t*)ws;
    float*  dmod = (float*)(ws + WBF_BYTES);
    bf16_t* xp   = (bf16_t*)(ws + WBF_BYTES + DMOD_BYTES);

    int n16 = (int)(XP_BYTES / 16);
    zero_ws_kernel<<<(n16 + 255) / 256, 256, 0, stream>>>((uint4*)xp, n16);
    wconv_kernel<<<WBF_ELEMS / 256, 256, 0, stream>>>(w, wbf);
    modnorm_kernel<<<COUT, 256, 0, stream>>>(w, y, dmod);
    xmod_transpose_kernel<<<dim3(CIN / 32, HW / 32, BATCH), 256, 0, stream>>>(x, y, xp);
    conv_gemm_kernel<<<dim3(COUT / 128, HW / 256, BATCH), 256, 0, stream>>>(wbf, xp, dmod, out);
}
